// Brain_46669114638464
// MI455X (gfx1250) — matrix operand order repaired and checked by reading, not compiled, not run on hardware
//
#include <hip/hip_runtime.h>
#include <hip/hip_bf16.h>

// ---------------------------------------------------------------------------
// Bidirectional LSTM (N=1024, T=512, H=128) for MI455X / gfx1250, wave32.
//   enc:  feat = enc2(relu(enc1(x)))                 -> bf16 in ws
//   lstm: gates = [x_t | h] @ [W_ih | W_hh]^T        -> WMMA bf16, f32 accum
//   dec:  out = dec2(relu(dec1(relu([h_fw|h_bw]))))  -> WMMA + wave reduce
// LSTM keeps the 256KB combined weight in LDS; each wave additionally holds
// half of its B fragments (128 VGPRs) register-resident across all 512 steps.
// x_t tiles stream in via GLOBAL_LOAD_ASYNC_TO_LDS_B128 (ASYNCcnt).
// ---------------------------------------------------------------------------

typedef __attribute__((ext_vector_type(16))) __bf16 v16bf;
typedef __attribute__((ext_vector_type(8)))  float  v8f;

#define NN   1024
#define TT   512
#define HH   128
#define NT   (NN * TT)          // 524288 rows

// ws layout (bytes)
#define WC_OFF    0u                     // 256 KB  combined [W_ih|W_hh] frags
#define E2_OFF    262144u                //  32 KB  enc2 frags
#define D1_OFF    294912u                //  64 KB  dec1 frags
#define FEAT_OFF  360448u                // 134 MB  feat bf16  [n][t][h]
#define HBUF_OFF  (360448u + 134217728u) // 268 MB  h bf16 [d][n][t][h]

union F16U { v16bf v; uint4 q[2]; };
union H8U  { __bf16 h[8]; uint4 q; };
union H16U { __bf16 h[16]; uint4 q[2]; };

__device__ __forceinline__ v16bf load_frag32B(const __bf16* p) {
    F16U f; f.q[0] = *(const uint4*)p; f.q[1] = *(const uint4*)(p + 8); return f.v;
}
// A fragment: halves 0..7 at k0, halves 8..15 at k0+16 (16-bit A layout)
__device__ __forceinline__ v16bf load_afrag(const __bf16* p) {
    F16U f; f.q[0] = *(const uint4*)p; f.q[1] = *(const uint4*)(p + 16); return f.v;
}
__device__ __forceinline__ float sigf(float x) { return 1.f / (1.f + __expf(-x)); }
__device__ __forceinline__ float tanh_fast(float x) {
    x = fminf(fmaxf(x, -15.f), 15.f);
    float e = __expf(2.f * x);
    return (e - 1.f) / (e + 1.f);
}
#define WMMA_BF16(a, b, c) __builtin_amdgcn_wmma_f32_16x16x32_bf16(false, (a), false, (b), (short)0, (c), false, false)

// gfx1250 async global->LDS copy (16B per lane), tracked by ASYNCcnt.
__device__ __forceinline__ void async_g2l_b128(unsigned lds_off, const void* gaddr) {
    asm volatile("global_load_async_to_lds_b128 %0, %1, off"
                 :: "v"(lds_off), "v"((unsigned long long)gaddr) : "memory");
}
__device__ __forceinline__ void wait_asynccnt0() {
    asm volatile("s_wait_asynccnt 0x0" ::: "memory");
}

// --------------------------- weight fragmentizers ---------------------------
// B-frag layout for tile (nt,kc): half value (k,g) with
//   k = kc*32 + (lane/16)*16 + i,  g = nt*16 + (lane%16)
// stored at dst[tile*512 + lane*16 + i]  (32 contiguous bytes per lane).

__global__ __launch_bounds__(256) void prep_wc(const float* __restrict__ wih,
                                               const float* __restrict__ whh,
                                               __bf16* __restrict__ dst) {
    int gid = blockIdx.x * 256 + threadIdx.x;          // 131072 halves
    if (gid >= 131072) return;
    int i = gid & 15, lane = (gid >> 4) & 31, tile = gid >> 9;
    int kc = tile & 7, nt = tile >> 3;
    int k = kc * 32 + ((lane >> 4) << 4) + i;
    int g = nt * 16 + (lane & 15);
    float v = (k < 128) ? wih[g * 128 + k] : whh[g * 128 + (k - 128)];
    dst[gid] = (__bf16)v;
}

__global__ __launch_bounds__(256) void prep_enc2(const float* __restrict__ w,
                                                 __bf16* __restrict__ dst) {
    int gid = blockIdx.x * 256 + threadIdx.x;          // 16384 halves
    if (gid >= 16384) return;
    int i = gid & 15, lane = (gid >> 4) & 31, tile = gid >> 9;   // tile = nt*4+kc
    int kc = tile & 3, nt = tile >> 2;
    int k = kc * 32 + ((lane >> 4) << 4) + i;
    int o = nt * 16 + (lane & 15);
    dst[gid] = (__bf16)w[o * 128 + k];
}

__global__ __launch_bounds__(256) void prep_dec1(const float* __restrict__ w,
                                                 __bf16* __restrict__ dst) {
    int gid = blockIdx.x * 256 + threadIdx.x;          // 32768 halves
    if (gid >= 32768) return;
    int i = gid & 15, lane = (gid >> 4) & 31, tile = gid >> 9;   // tile = nt*8+kc
    int kc = tile & 7, nt = tile >> 3;
    int k = kc * 32 + ((lane >> 4) << 4) + i;
    int o = nt * 16 + (lane & 15);
    dst[gid] = (__bf16)w[o * 256 + k];
}

// -------------------------------- encoder ----------------------------------
// One block per 16 flattened (n,t) rows. enc1 (K=2) on VALU; enc2 via WMMA.
__global__ __launch_bounds__(256) void encoder_kernel(
    const float* __restrict__ x, const float* __restrict__ e1w,
    const float* __restrict__ e1b, const float* __restrict__ e2b,
    const __bf16* __restrict__ e2frag, __bf16* __restrict__ feat) {
    __shared__ __align__(16) __bf16 As[16 * 128];
    __shared__ __align__(16) float  Os[16 * 128];
    const int tid = threadIdx.x;
    const size_t rowbase = (size_t)blockIdx.x * 16;

    { // stage feat1 = relu(enc1(x)) as bf16, WMMA-A row-major
        int row = tid >> 4, h0 = (tid & 15) * 8;
        size_t ridx = rowbase + row;
        float x0 = x[ridx * 2], x1 = x[ridx * 2 + 1];
        H8U o;
#pragma unroll
        for (int j = 0; j < 8; ++j) {
            int h = h0 + j;
            float v = fmaf(e1w[h * 2], x0, fmaf(e1w[h * 2 + 1], x1, e1b[h]));
            o.h[j] = (__bf16)fmaxf(v, 0.f);
        }
        *(uint4*)(&As[row * 128 + h0]) = o.q;
    }
    __syncthreads();

    { // enc2: K=128, wave w owns output cols [w*16, w*16+16)
        const int lane = tid & 31, wave = tid >> 5, grp = lane >> 4, m = lane & 15;
        v8f acc = {0.f, 0.f, 0.f, 0.f, 0.f, 0.f, 0.f, 0.f};
#pragma unroll
        for (int kc = 0; kc < 4; ++kc) {
            v16bf a = load_frag32B(&As[m * 128 + kc * 32 + grp * 16]);
            v16bf b = load_frag32B(e2frag + ((size_t)(wave * 4 + kc) * 32 + lane) * 16);
            acc = WMMA_BF16(a, b, acc);
        }
        int col = wave * 16 + m;
        float bias = e2b[col];
#pragma unroll
        for (int r = 0; r < 8; ++r)
            Os[(r + 8 * grp) * 128 + col] = acc[r] + bias;
    }
    __syncthreads();

    { // coalesced bf16 writeback
        int row = tid >> 4, h0 = (tid & 15) * 8;
        H8U o;
#pragma unroll
        for (int j = 0; j < 8; ++j) o.h[j] = (__bf16)Os[row * 128 + h0 + j];
        *(uint4*)(feat + (rowbase + row) * 128 + h0) = o.q;
    }
}

// --------------------------------- LSTM ------------------------------------
// block = (dir, 16-row batch tile). Weights (256KB) resident in LDS; half of
// each wave's B fragments additionally pinned in VGPRs across all steps.
__global__ __launch_bounds__(256) void lstm_kernel(
    const __bf16* __restrict__ feat, const __bf16* __restrict__ wcfrag,
    __bf16* __restrict__ hbuf) {
    __shared__ __align__(16) __bf16 WfS[131072];   // 256 KB combined W frags
    __shared__ __align__(16) __bf16 As[16 * 256];  // [x_t | h] bf16
    __shared__ __align__(16) float  Gs[16 * 512];  // gates f32
    __shared__ __align__(16) float  Cs[16 * 128];  // cell state f32

    const int tid  = threadIdx.x;
    const int d    = blockIdx.x >> 6;
    const int nb   = (blockIdx.x & 63) * 16;
    const int lane = tid & 31, wave = tid >> 5, grp = lane >> 4, m = lane & 15;
    const int row  = tid >> 4, ch = tid & 15, h0 = ch * 8;
    const size_t n = nb + row;

    for (int j = tid; j < 131072 / 8; j += 256)
        ((uint4*)WfS)[j] = ((const uint4*)wcfrag)[j];
    for (int j = tid; j < 16 * 128; j += 256) Cs[j] = 0.f;
    for (int j = tid; j < 16 * 256; j += 256) As[j] = (__bf16)0.f;
    __syncthreads();

    // pin B tiles j=0,1 (all 8 k-chunks) in registers for the whole sequence
    v16bf Breg[16];
#pragma unroll
    for (int jk = 0; jk < 16; ++jk) {
        int j = jk >> 3, kc = jk & 7;
        Breg[jk] = load_frag32B(WfS + (((wave * 4 + j) * 8 + kc) * 32 + lane) * 16);
    }

    const unsigned lds_x = (unsigned)(size_t)(&As[row * 256 + ch * 8]);

    for (int t = 0; t < TT; ++t) {
        const int tp = d ? (TT - 1 - t) : t;
        // async-stage x_t tile (16 x 128 bf16) straight into A[:,0:128]
        async_g2l_b128(lds_x, feat + (n * TT + tp) * 128 + ch * 8);
        if (t + 1 < TT) {
            const int tn = d ? (TT - 2 - t) : (t + 1);
            __builtin_prefetch(feat + (n * TT + tn) * 128 + ch * 8, 0, 1);
        }
        wait_asynccnt0();
        __syncthreads();

        // gates(16x512) = [x_t|h](16x256) @ Wc^T : 8 k-chunks, 4 col tiles/wave
        v8f acc[4];
#pragma unroll
        for (int j = 0; j < 4; ++j) acc[j] = (v8f){0.f,0.f,0.f,0.f,0.f,0.f,0.f,0.f};
#pragma unroll
        for (int kc = 0; kc < 8; ++kc) {
            v16bf a = load_frag32B(&As[m * 256 + kc * 32 + grp * 16]);
            acc[0] = WMMA_BF16(a, Breg[kc],     acc[0]);
            acc[1] = WMMA_BF16(a, Breg[8 + kc], acc[1]);
            v16bf b2 = load_frag32B(WfS + (((wave * 4 + 2) * 8 + kc) * 32 + lane) * 16);
            acc[2] = WMMA_BF16(a, b2, acc[2]);
            v16bf b3 = load_frag32B(WfS + (((wave * 4 + 3) * 8 + kc) * 32 + lane) * 16);
            acc[3] = WMMA_BF16(a, b3, acc[3]);
        }
#pragma unroll
        for (int j = 0; j < 4; ++j) {
            int colg = (wave * 4 + j) * 16 + m;
#pragma unroll
            for (int r = 0; r < 8; ++r)
                Gs[(r + 8 * grp) * 512 + colg] = acc[j][r];
        }
        __syncthreads();

        // elementwise cell update: thread owns (row, h0..h0+7) forever
        H8U hb;
#pragma unroll
        for (int j = 0; j < 8; ++j) {
            int h = h0 + j;
            float ig = Gs[row * 512 + h];
            float fg = Gs[row * 512 + 128 + h];
            float gg = Gs[row * 512 + 256 + h];
            float og = Gs[row * 512 + 384 + h];
            float c  = sigf(fg) * Cs[row * 128 + h] + sigf(ig) * tanh_fast(gg);
            Cs[row * 128 + h] = c;
            hb.h[j] = (__bf16)(sigf(og) * tanh_fast(c));
        }
        *(uint4*)(&As[row * 256 + 128 + h0]) = hb.q;                     // h -> A
        *(uint4*)(hbuf + (((size_t)d * NN + n) * TT + tp) * 128 + h0) = hb.q;
        // next x-stage races nothing; barrier at loop top orders h vs WMMA
    }
}

// -------------------------------- decoder ----------------------------------
__global__ __launch_bounds__(256) void decoder_kernel(
    const __bf16* __restrict__ hbuf, const __bf16* __restrict__ d1frag,
    const float* __restrict__ d1b, const float* __restrict__ d2w,
    const float* __restrict__ d2b, float* __restrict__ out) {
    __shared__ __align__(16) __bf16 As[16 * 256];   // relu([h_fw|h_bw])
    __shared__ __align__(16) float  Ps[8 * 16];     // per-wave row partials
    const int tid = threadIdx.x;
    const size_t rowbase = (size_t)blockIdx.x * 16;

    { // stage with ReLU
        int row = tid >> 4, ch = tid & 15;
        size_t ridx = rowbase + row;
        const __bf16* src = (ch < 8)
            ? hbuf + ridx * 128 + ch * 16
            : hbuf + ((size_t)NT + ridx) * 128 + (ch - 8) * 16;
        H16U v;
        v.q[0] = *(const uint4*)src; v.q[1] = *(const uint4*)(src + 8);
#pragma unroll
        for (int j = 0; j < 16; ++j) v.h[j] = (__bf16)fmaxf((float)v.h[j], 0.f);
        *(uint4*)(&As[row * 256 + ch * 16])     = v.q[0];
        *(uint4*)(&As[row * 256 + ch * 16 + 8]) = v.q[1];
    }
    __syncthreads();

    { // q = relu(dec1(.)+b1); out = q . dec2_w + b2 via shuffle-tree reduce
        const int lane = tid & 31, wave = tid >> 5, grp = lane >> 4, m = lane & 15;
        v8f acc = {0.f, 0.f, 0.f, 0.f, 0.f, 0.f, 0.f, 0.f};
#pragma unroll
        for (int kc = 0; kc < 8; ++kc) {
            v16bf a = load_frag32B(&As[m * 256 + kc * 32 + grp * 16]);
            v16bf b = load_frag32B(d1frag + ((size_t)(wave * 8 + kc) * 32 + lane) * 16);
            acc = WMMA_BF16(a, b, acc);
        }
        int col = wave * 16 + m;
        float bias = d1b[col], w2 = d2w[col];
#pragma unroll
        for (int r = 0; r < 8; ++r) {
            float v = fmaxf(acc[r] + bias, 0.f) * w2;
#pragma unroll
            for (int off = 8; off > 0; off >>= 1) v += __shfl_xor(v, off, 32);
            if (m == 0) Ps[wave * 16 + r + 8 * grp] = v;   // deterministic tree
        }
    }
    __syncthreads();
    if (tid < 16) {
        float s = d2b[0];
#pragma unroll
        for (int w = 0; w < 8; ++w) s += Ps[w * 16 + tid];
        out[rowbase + tid] = s;
    }
}

// -------------------------------- launcher ---------------------------------
extern "C" void kernel_launch(void* const* d_in, const int* in_sizes, int n_in,
                              void* d_out, int out_size, void* d_ws, size_t ws_size,
                              hipStream_t stream) {
    const float* x   = (const float*)d_in[0];
    const float* e1w = (const float*)d_in[1];
    const float* e1b = (const float*)d_in[2];
    const float* e2w = (const float*)d_in[3];
    const float* e2b = (const float*)d_in[4];
    const float* wih = (const float*)d_in[5];
    const float* whh = (const float*)d_in[6];
    const float* d1w = (const float*)d_in[7];
    const float* d1b = (const float*)d_in[8];
    const float* d2w = (const float*)d_in[9];
    const float* d2b = (const float*)d_in[10];

    unsigned char* ws = (unsigned char*)d_ws;
    __bf16* wcf  = (__bf16*)(ws + WC_OFF);
    __bf16* e2f  = (__bf16*)(ws + E2_OFF);
    __bf16* d1f  = (__bf16*)(ws + D1_OFF);
    __bf16* feat = (__bf16*)(ws + FEAT_OFF);
    __bf16* hbuf = (__bf16*)(ws + HBUF_OFF);

    prep_wc  <<<512, 256, 0, stream>>>(wih, whh, wcf);
    prep_enc2<<< 64, 256, 0, stream>>>(e2w, e2f);
    prep_dec1<<<128, 256, 0, stream>>>(d1w, d1f);

    encoder_kernel<<<NT / 16, 256, 0, stream>>>(x, e1w, e1b, e2b, e2f, feat);
    lstm_kernel   <<<128,     256, 0, stream>>>(feat, wcf, hbuf);
    decoder_kernel<<<NT / 16, 256, 0, stream>>>(hbuf, d1f, d1b, d2w, d2b, (float*)d_out);
}
